// Samba_15702400434631
// MI455X (gfx1250) — matrix operand order repaired and checked by reading, not compiled, not run on hardware
//
#include <hip/hip_runtime.h>

// ---------------------------------------------------------------------------
// Samba (4-layer Mamba + tied LM head + 3-layer readout MLP) forward, gfx1250.
// GEMMs run on v_wmma_f32_16x16x32_bf16 (fp32 weights converted to bf16 in
// registers/LDS, fp32 accumulation). Workload is HBM-bound (~440MB weights,
// ~60 GFLOP): double-buffered LDS + register prefetch pipelines HBM behind
// WMMA; split-K spreads the 201MB readout GEMM across the whole chip.
// ---------------------------------------------------------------------------

typedef __bf16 bf16;
typedef __attribute__((ext_vector_type(16))) __bf16 v16bf;
typedef __attribute__((ext_vector_type(8)))  __bf16 v8bf;
typedef __attribute__((ext_vector_type(4)))  __bf16 v4bf;
typedef __attribute__((ext_vector_type(8)))  float  v8f;
typedef __attribute__((ext_vector_type(4)))  float  v4f;

#define V_    50280
#define DM_   768
#define NL_   4
#define DS_   16
#define DC_   4
#define DTR_  48
#define ROH_  512
#define DI_   1536
#define B_    2
#define L_    128
#define ROWS_ (B_ * L_)            // 256
#define CHUNK_ (DI_ * DS_)         // 24576 (per-layer feature chunk)
#define LSTRIDE_ ((size_t)ROWS_ * CHUNK_)  // 6291456 (layer stride in hidden)

#define BM 64
#define BN 128
#define BK 64
#define BKP (BK + 8)               // padded LDS row (bf16 units); 144B, 16B-aligned
#define NAQ ((BM * BK) / (256 * 4))   // A prefetch quads per thread = 4
#define NBQ ((BK * BN) / (256 * 4))   // B prefetch quads per thread = 8

__device__ __forceinline__ v16bf frag_from2(const bf16* p0, const bf16* p1) {
    union { v16bf v; v8bf h[2]; } u;
    u.h[0] = *(const v8bf*)p0;
    u.h[1] = *(const v8bf*)(p1);
    return u.v;
}

// ---------------------------------------------------------------------------
// C[M,N] (+)= A[M,K] * B[K,N]
//   AMODE: 0 = row-major A (stride lda)
//          1 = FEAT: A(row,k) = hidden[k/CHUNK][row][k%CHUNK]  (readout feats)
//   BT   : 0 = B[k*ldb+n], 1 = B[n*ldb+k]   (tied lm head: embed.T)
//   split-K via blockIdx.z: slice [z*ksplit, min(K,(z+1)*ksplit)), output
//   slice z stored at C + z*M*ldc (z=0 for unsplit launches).
// ---------------------------------------------------------------------------
template <int AMODE, int BT>
__global__ __launch_bounds__(256) void gemm_wmma(
    const float* __restrict__ A, int lda,
    const float* __restrict__ Bm, int ldb,
    float* __restrict__ C, int ldc,
    int M, int N, int K, int ksplit, int accum)
{
    __shared__ __align__(16) bf16 As[2][BM][BKP];
    __shared__ __align__(16) bf16 Bs[2][BN][BKP];  // n-major: contiguous frags

    const int tid  = threadIdx.x;
    const int lane = tid & 31;
    const int wave = tid >> 5;         // 8 waves
    const int wm   = wave >> 2;        // 0..1 (32-row tile)
    const int wn   = wave & 3;         // 0..3 (32-col tile)
    const int r0   = blockIdx.y * BM;
    const int n0   = blockIdx.x * BN;
    const int kbeg = blockIdx.z * ksplit;
    const int kend = (kbeg + ksplit < K) ? (kbeg + ksplit) : K;
    C += (size_t)blockIdx.z * (size_t)M * (size_t)ldc;

    v8f acc[2][2];
    const v8f zf = {0.f,0.f,0.f,0.f,0.f,0.f,0.f,0.f};
#pragma unroll
    for (int i = 0; i < 2; ++i)
#pragma unroll
        for (int j = 0; j < 2; ++j) acc[i][j] = zf;

    const int hs8  = (lane & 16) ? 8 : 0;   // A-frag K half-select
    const int hs16 = (lane & 16) ? 16 : 0;  // unused (B frags also use hs8)
    const int l15  = lane & 15;

    v4f areg[NAQ], breg[NBQ];

    // ---- global -> register prefetch of one k-tile ----
    auto load_tile = [&](int k0) {
#pragma unroll
        for (int it = 0; it < NAQ; ++it) {
            int q  = tid + it * 256;
            int ar = q >> 4;               // BK/4 = 16 quads per row
            int aq = (q & 15) << 2;
            int gr = r0 + ar, gk = k0 + aq;
            v4f v = {0.f, 0.f, 0.f, 0.f};
            if (gr < M && gk < kend) {
                if (AMODE == 0) {
                    v = *(const v4f*)(A + (size_t)gr * lda + gk);
                } else {
                    int li = gk / CHUNK_;
                    int kk = gk - li * CHUNK_;
                    v = *(const v4f*)(A + (size_t)li * LSTRIDE_ +
                                      (size_t)gr * CHUNK_ + kk);
                }
            }
            areg[it] = v;
        }
#pragma unroll
        for (int it = 0; it < NBQ; ++it) {
            int q = tid + it * 256;
            v4f v = {0.f, 0.f, 0.f, 0.f};
            if (BT == 0) {
                int bk = q >> 5;           // BN/4 = 32 quads per k-row
                int bn = (q & 31) << 2;
                int gk = k0 + bk, gn = n0 + bn;
                if (gk < kend && gn < N)
                    v = *(const v4f*)(Bm + (size_t)gk * ldb + gn);
            } else {
                int bn = q >> 4;           // BK/4 = 16 quads per n-row
                int bq = (q & 15) << 2;
                int gn = n0 + bn, gk = k0 + bq;
                if (gn < N && gk < kend)
                    v = *(const v4f*)(Bm + (size_t)gn * ldb + gk);
            }
            breg[it] = v;
        }
    };

    // ---- register -> LDS (fp32 -> bf16) ----
    auto store_tile = [&](int buf) {
#pragma unroll
        for (int it = 0; it < NAQ; ++it) {
            int q  = tid + it * 256;
            int ar = q >> 4;
            int aq = (q & 15) << 2;
            v4bf pk;
#pragma unroll
            for (int j = 0; j < 4; ++j) pk[j] = (bf16)areg[it][j];
            *(v4bf*)&As[buf][ar][aq] = pk;
        }
#pragma unroll
        for (int it = 0; it < NBQ; ++it) {
            int q = tid + it * 256;
            if (BT == 0) {
                int bk = q >> 5;
                int bn = (q & 31) << 2;
#pragma unroll
                for (int j = 0; j < 4; ++j) Bs[buf][bn + j][bk] = (bf16)breg[it][j];
            } else {
                int bn = q >> 4;
                int bq = (q & 15) << 2;
                v4bf pk;
#pragma unroll
                for (int j = 0; j < 4; ++j) pk[j] = (bf16)breg[it][j];
                *(v4bf*)&Bs[buf][bn][bq] = pk;
            }
        }
    };

    // ---- 8 WMMAs (2 K=32 sub-steps x 2x2 frags) from LDS buffer ----
    auto compute = [&](int buf) {
#pragma unroll
        for (int ks = 0; ks < 2; ++ks) {
            const int kb = ks * 32;
            v16bf af[2], bfr[2];
#pragma unroll
            for (int i = 0; i < 2; ++i) {
                const bf16* row = &As[buf][wm * 32 + i * 16 + l15][kb + hs8];
                af[i] = frag_from2(row, row + 16);   // K 0-7 / 16-23 (hs: +8)
            }
#pragma unroll
            for (int j = 0; j < 2; ++j) {
                const bf16* row = &Bs[buf][wn * 32 + j * 16 + l15][kb + hs8];
                bfr[j] = frag_from2(row, row + 16);   // K 0-7 / 16-23 (hs: +8)
            }
#pragma unroll
            for (int i = 0; i < 2; ++i)
#pragma unroll
                for (int j = 0; j < 2; ++j)
                    acc[i][j] = __builtin_amdgcn_wmma_f32_16x16x32_bf16(
                        false, af[i], false, bfr[j],
                        (short)0, acc[i][j], false, false);
        }
    };

    // ---- software-pipelined k loop: 1 barrier / tile, HBM behind WMMA ----
    const int niters = (kend - kbeg + BK - 1) / BK;
    load_tile(kbeg);
    store_tile(0);
    __syncthreads();
    for (int i = 0; i < niters; ++i) {
        const bool have_next = (i + 1) < niters;
        if (have_next) load_tile(kbeg + (i + 1) * BK);  // issue HBM early
        compute(i & 1);                                 // WMMA on current buf
        if (have_next) store_tile((i + 1) & 1);         // fill other buf
        __syncthreads();
    }

    // ---- store D (VGPR r: row = r + 8*laneHalf, col = lane&15) ----
#pragma unroll
    for (int i = 0; i < 2; ++i) {
#pragma unroll
        for (int j = 0; j < 2; ++j) {
            const int col = n0 + wn * 32 + j * 16 + l15;
#pragma unroll
            for (int r = 0; r < 8; ++r) {
                const int row = r0 + wm * 32 + i * 16 + r + ((lane & 16) ? 8 : 0);
                if (row < M && col < N) {
                    size_t idx = (size_t)row * ldc + col;
                    float v = acc[i][j][r];
                    if (accum) v += C[idx];
                    C[idx] = v;
                }
            }
        }
    }
}

// ---------------------------------------------------------------------------
// split-K reduction with fused bias + activation (mode 1 = relu)
// ---------------------------------------------------------------------------
__global__ void reduce_kernel(const float* __restrict__ part, int S, int MN,
                              const float* __restrict__ bias, int ncol,
                              float* __restrict__ out, int mode)
{
    int t = blockIdx.x * blockDim.x + threadIdx.x;
    if (t >= MN) return;
    float s = 0.f;
    for (int z = 0; z < S; ++z) s += part[(size_t)z * MN + t];
    s += bias[t % ncol];
    if (mode == 1) s = fmaxf(s, 0.f);
    out[t] = s;
}

// ---------------------------------------------------------------------------
// Elementwise / small kernels
// ---------------------------------------------------------------------------
__global__ void embed_kernel(const int* __restrict__ ids,
                             const float* __restrict__ emb,
                             float* __restrict__ x)
{
    int t = blockIdx.x * blockDim.x + threadIdx.x;
    if (t >= ROWS_ * DM_) return;
    int row = t / DM_, d = t - row * DM_;
    x[t] = emb[(size_t)ids[row] * DM_ + d];
}

__global__ __launch_bounds__(256) void rmsnorm_kernel(
    const float* __restrict__ x, const float* __restrict__ w,
    float* __restrict__ out)
{
    __shared__ float red[256];
    const int r = blockIdx.x, t = threadIdx.x;
    float s = 0.f;
    for (int d = t; d < DM_; d += 256) {
        float v = x[(size_t)r * DM_ + d];
        s += v * v;
    }
    red[t] = s;
    __syncthreads();
    for (int off = 128; off > 0; off >>= 1) {
        if (t < off) red[t] += red[t + off];
        __syncthreads();
    }
    const float rs = rsqrtf(red[0] / (float)DM_ + 1e-5f);
    for (int d = t; d < DM_; d += 256)
        out[(size_t)r * DM_ + d] = x[(size_t)r * DM_ + d] * rs * w[d];
}

// causal depthwise conv (DC=4) over the xc half of xz, then SiLU
__global__ void conv_silu_kernel(const float* __restrict__ xz,
                                 const float* __restrict__ cw,
                                 const float* __restrict__ cb,
                                 float* __restrict__ xconv)
{
    int t = blockIdx.x * blockDim.x + threadIdx.x;
    if (t >= ROWS_ * DI_) return;
    int row = t / DI_, d = t - row * DI_;
    int b = row / L_, l = row - b * L_;
    float acc = cb[d];
#pragma unroll
    for (int k = 0; k < DC_; ++k) {
        int li = l + k - (DC_ - 1);
        if (li >= 0)
            acc += cw[d * DC_ + k] * xz[(size_t)(b * L_ + li) * (2 * DI_) + d];
    }
    xconv[t] = acc / (1.f + expf(-acc));   // silu
}

// mode: 0 = +bias, 1 = relu(+bias), 2 = softplus(+bias)
__global__ void biasact_kernel(float* __restrict__ C,
                               const float* __restrict__ bias,
                               int ncol, int total, int mode)
{
    int t = blockIdx.x * blockDim.x + threadIdx.x;
    if (t >= total) return;
    float v = C[t] + bias[t % ncol];
    if (mode == 1)      v = fmaxf(v, 0.f);
    else if (mode == 2) v = (v > 20.f) ? v : log1pf(expf(v));
    C[t] = v;
}

// selective scan: one thread per (batch, channel); 16 SSM states in registers.
__global__ void scan_kernel(const float* __restrict__ delta,
                            const float* __restrict__ dbc,
                            const float* __restrict__ xconv,
                            const float* __restrict__ A_log_l,
                            float* __restrict__ hidden_l,
                            float* __restrict__ yscan)
{
    int t = blockIdx.x * blockDim.x + threadIdx.x;
    if (t >= B_ * DI_) return;
    int b = t / DI_, d = t - b * DI_;
    float Arow[DS_], h[DS_];
#pragma unroll
    for (int s = 0; s < DS_; ++s) {
        Arow[s] = -expf(A_log_l[d * DS_ + s]);
        h[s] = 0.f;
    }
    for (int l = 0; l < L_; ++l) {
        int row = b * L_ + l;
        float dlt = delta[(size_t)row * DI_ + d];
        float xc  = xconv[(size_t)row * DI_ + d];
        float y = 0.f;
        float* hout = hidden_l + (size_t)row * CHUNK_ + (size_t)d * DS_;
#pragma unroll
        for (int s = 0; s < DS_; ++s) {
            float Bv = dbc[(size_t)row * 80 + DTR_ + s];
            float Cv = dbc[(size_t)row * 80 + DTR_ + DS_ + s];
            float hn = expf(dlt * Arow[s]) * h[s] + dlt * Bv * xc;
            h[s] = hn;
            hout[s] = hn;
            y += hn * Cv;
        }
        yscan[(size_t)row * DI_ + d] = y;
    }
}

// y = (yscan + D*xconv) * silu(z)
__global__ void gate_kernel(const float* __restrict__ ys,
                            const float* __restrict__ xconv,
                            const float* __restrict__ xz,
                            const float* __restrict__ Dp,
                            float* __restrict__ yg)
{
    int t = blockIdx.x * blockDim.x + threadIdx.x;
    if (t >= ROWS_ * DI_) return;
    int row = t / DI_, d = t - row * DI_;
    float z = xz[(size_t)row * (2 * DI_) + DI_ + d];
    float sz = z / (1.f + expf(-z));
    yg[t] = (ys[t] + Dp[d] * xconv[t]) * sz;
}

// ---------------------------------------------------------------------------
// Host orchestration
// ---------------------------------------------------------------------------
extern "C" void kernel_launch(void* const* d_in, const int* in_sizes, int n_in,
                              void* d_out, int out_size, void* d_ws, size_t ws_size,
                              hipStream_t stream)
{
    const int*   ids       = (const int*)  d_in[0];
    const float* embed     = (const float*)d_in[1];
    const float* norm_f    = (const float*)d_in[2];
    const float* norm_w    = (const float*)d_in[3];
    const float* in_proj_w = (const float*)d_in[4];
    const float* conv_w    = (const float*)d_in[5];
    const float* conv_b    = (const float*)d_in[6];
    const float* x_proj_w  = (const float*)d_in[7];
    const float* dt_w      = (const float*)d_in[8];
    const float* dt_b      = (const float*)d_in[9];
    const float* A_log     = (const float*)d_in[10];
    const float* Dp        = (const float*)d_in[11];
    const float* out_projw = (const float*)d_in[12];
    const float* ro_w1     = (const float*)d_in[13];
    const float* ro_b1     = (const float*)d_in[14];
    const float* ro_w2     = (const float*)d_in[15];
    const float* ro_b2     = (const float*)d_in[16];
    const float* ro_w3     = (const float*)d_in[17];
    const float* ro_b3     = (const float*)d_in[18];

    float* out      = (float*)d_out;
    float* main_lg  = out;                              // (256, V)
    float* read_lg  = out + (size_t)ROWS_ * V_;         // (256, V)
    float* hidden   = out + 2 * (size_t)ROWS_ * V_;     // (NL, 256, DI, DS)

    // workspace carve-up (~19.3 MB fp32)
    const int SK1 = 16;                                 // split-K ways for ro1
    float* p = (float*)d_ws;
    float* x     = p; p += (size_t)ROWS_ * DM_;
    float* xn    = p; p += (size_t)ROWS_ * DM_;
    float* xz    = p; p += (size_t)ROWS_ * 2 * DI_;
    float* xconv = p; p += (size_t)ROWS_ * DI_;
    float* dbc   = p; p += (size_t)ROWS_ * 80;
    float* delta = p; p += (size_t)ROWS_ * DI_;
    float* ysc   = p; p += (size_t)ROWS_ * DI_;
    float* yg    = p; p += (size_t)ROWS_ * DI_;
    float* h1    = p; p += (size_t)ROWS_ * ROH_;
    float* h2    = p; p += (size_t)ROWS_ * (ROH_ / 2);
    float* part  = p; p += (size_t)SK1 * ROWS_ * ROH_;  // 8 MB partials

    auto blocks = [](size_t n) { return dim3((unsigned)((n + 255) / 256)); };
    auto gemm_nn = [&](const float* A, int lda, const float* Bp, int ldb,
                       float* Cp, int ldc, int M, int N, int K, int acc) {
        dim3 grid((N + BN - 1) / BN, (M + BM - 1) / BM, 1);
        gemm_wmma<0, 0><<<grid, 256, 0, stream>>>(A, lda, Bp, ldb, Cp, ldc,
                                                  M, N, K, K, acc);
    };

    // embedding gather
    embed_kernel<<<blocks((size_t)ROWS_ * DM_), 256, 0, stream>>>(ids, embed, x);

    for (int l = 0; l < NL_; ++l) {
        rmsnorm_kernel<<<ROWS_, 256, 0, stream>>>(x, norm_w + (size_t)l * DM_, xn);
        // in_proj: (256,768)x(768,3072)
        gemm_nn(xn, DM_, in_proj_w + (size_t)l * DM_ * 2 * DI_, 2 * DI_,
                xz, 2 * DI_, ROWS_, 2 * DI_, DM_, 0);
        conv_silu_kernel<<<blocks((size_t)ROWS_ * DI_), 256, 0, stream>>>(
            xz, conv_w + (size_t)l * DI_ * DC_, conv_b + (size_t)l * DI_, xconv);
        // x_proj: (256,1536)x(1536,80)
        gemm_nn(xconv, DI_, x_proj_w + (size_t)l * DI_ * 80, 80,
                dbc, 80, ROWS_, 80, DI_, 0);
        // delta = softplus(dt @ dt_w + dt_b): (256,48)x(48,1536)
        gemm_nn(dbc, 80, dt_w + (size_t)l * DTR_ * DI_, DI_,
                delta, DI_, ROWS_, DI_, DTR_, 0);
        biasact_kernel<<<blocks((size_t)ROWS_ * DI_), 256, 0, stream>>>(
            delta, dt_b + (size_t)l * DI_, DI_, ROWS_ * DI_, 2);
        // selective scan (writes hs to output hidden region)
        scan_kernel<<<blocks((size_t)B_ * DI_), 256, 0, stream>>>(
            delta, dbc, xconv, A_log + (size_t)l * DI_ * DS_,
            hidden + (size_t)l * LSTRIDE_, ysc);
        gate_kernel<<<blocks((size_t)ROWS_ * DI_), 256, 0, stream>>>(
            ysc, xconv, xz, Dp + (size_t)l * DI_, yg);
        // out_proj with fused residual: x += yg @ out_proj_w
        gemm_nn(yg, DI_, out_projw + (size_t)l * DI_ * DM_, DM_,
                x, DM_, ROWS_, DM_, DI_, 1);
    }

    // final norm + tied lm head: main_logits = xf @ embed.T  (BT variant)
    rmsnorm_kernel<<<ROWS_, 256, 0, stream>>>(x, norm_f, xn);
    {
        dim3 grid((V_ + BN - 1) / BN, (ROWS_ + BM - 1) / BM, 1);
        gemm_wmma<0, 1><<<grid, 256, 0, stream>>>(xn, DM_, embed, DM_,
                                                  main_lg, V_,
                                                  ROWS_, V_, DM_, DM_, 0);
    }

    // readout layer 1: FEAT A-addressing, K=98304, split-K x16 (201MB weight
    // read spread over 256 workgroups), then deterministic reduce+bias+relu.
    {
        const int K1 = NL_ * CHUNK_;             // 98304
        dim3 grid((ROH_ + BN - 1) / BN, (ROWS_ + BM - 1) / BM, SK1);
        gemm_wmma<1, 0><<<grid, 256, 0, stream>>>(hidden, 0, ro_w1, ROH_,
                                                  part, ROH_,
                                                  ROWS_, ROH_, K1, K1 / SK1, 0);
        reduce_kernel<<<blocks((size_t)ROWS_ * ROH_), 256, 0, stream>>>(
            part, SK1, ROWS_ * ROH_, ro_b1, ROH_, h1, 1);
    }
    // readout layer 2
    gemm_nn(h1, ROH_, ro_w2, ROH_ / 2, h2, ROH_ / 2, ROWS_, ROH_ / 2, ROH_, 0);
    biasact_kernel<<<blocks((size_t)ROWS_ * (ROH_ / 2)), 256, 0, stream>>>(
        h2, ro_b2, ROH_ / 2, ROWS_ * (ROH_ / 2), 1);
    // readout layer 3 -> logits
    gemm_nn(h2, ROH_ / 2, ro_w3, V_, read_lg, V_, ROWS_, V_, ROH_ / 2, 0);
    biasact_kernel<<<blocks((size_t)ROWS_ * V_), 256, 0, stream>>>(
        read_lg, ro_b3, V_, ROWS_ * V_, 0);
}
